// ChebyNet_39737037422834
// MI455X (gfx1250) — matrix operand order repaired and checked by reading, not compiled, not run on hardware
//
#include <hip/hip_runtime.h>
#include <hip/hip_bf16.h>
#include <math.h>

// ---------------- problem constants (match reference) ----------------
constexpr int B = 8, N = 1024, NTYPES = 2, MN = 128, M = NTYPES * MN;
constexpr int BETA = 8, M1 = 16, M2 = 4, NFEAT = M1 * M2;   // 64
constexpr int H = 128, NGHOST = 64;
constexpr float RMAX = 6.0f, RMIN = 0.5f, SPAN = RMAX - RMIN;
constexpr float PI_F = 3.14159265358979323846f;

// ---------------- output layout (floats, concatenated) ----------------
constexpr int OUT_ETOT  = 0;                                   // (B,1)      = 8
constexpr int OUT_EI    = OUT_ETOT + B;                        // (B,N)      = 8192
constexpr int OUT_FORCE = OUT_EI + B * N;                      // (B,N+G,3)  = 26112
constexpr int OUT_VIR   = OUT_FORCE + B * (N + NGHOST) * 3;    // (B,9)      = 72
constexpr int OUT_TOTAL = OUT_VIR + B * 9;                     // 34384

// ---------------- workspace layout (floats) ----------------
constexpr size_t WS_FEAT  = 0;                                  // B*N*64
constexpr size_t WS_S     = WS_FEAT  + (size_t)B * N * NFEAT;   // B*N*64  (S[p][a])
constexpr size_t WS_H0    = WS_S     + (size_t)B * N * NFEAT;   // B*N*128
constexpr size_t WS_T1    = WS_H0    + (size_t)B * N * H;       // B*N*128
constexpr size_t WS_T2    = WS_T1    + (size_t)B * N * H;       // B*N*128
constexpr size_t WS_DFEAT = WS_T2    + (size_t)B * N * H;       // B*N*64
constexpr size_t WS_DS    = WS_DFEAT + (size_t)B * N * NFEAT;   // B*N*64
constexpr size_t WS_STATS = WS_DS    + (size_t)B * N * NFEAT;   // 8: sum[2],sumsq[2],mean[2],std[2]
// total ~5.25M floats (~21 MB)

// ---------------- async-to-LDS (gfx1250 ASYNCcnt path), compile-safe guard ----------------
#if defined(__has_builtin)
#if __has_builtin(__builtin_amdgcn_global_load_async_to_lds_b128) && \
    __has_builtin(__builtin_amdgcn_s_wait_asynccnt)
#define HAVE_ASYNC_LDS 1
#endif
#endif

#ifdef HAVE_ASYNC_LDS
// Builtin signature (from hipcc diagnostic): first param is
// '__attribute__((vector_size(16))) int __device__ *' i.e. non-const int4* in AS(1).
typedef int v4i_t __attribute__((vector_size(16)));
#define GPTR(p) ((__attribute__((address_space(1))) v4i_t*)(p))
#define LPTR(p) ((__attribute__((address_space(3))) v4i_t*)(p))
#endif

// ---------------- WMMA types & helpers (CDNA5 gfx1250, wave32) ----------------
typedef __attribute__((ext_vector_type(16))) _Float16 v16h;
typedef __attribute__((ext_vector_type(8)))  float    v8f;

// A-matrix 16x32 f16 layout: lane L row = L&15;
// K = kb + 16*(L>>4) + i (contiguous, matches B loaders).
__device__ inline v16h lds_load_A(const float* X, int ld, int lane, int kb) {
  const float* row = X + (size_t)(lane & 15) * ld;
  const int khi = (lane >> 4) << 4;
  v16h a;
#pragma unroll
  for (int i = 0; i < 16; ++i) {
    const int k = kb + i + khi;
    a[i] = (_Float16)row[k];
  }
  return a;
}

// B-matrix 32x16 f16 layout: lanes 0-15 col=lane, K = kb + i; lanes 16-31 col=lane-16, K = kb + 16 + i.
// W is row-major [K][ldn].
__device__ inline v16h glb_load_B(const float* W, int ldn, int lane, int kb, int nb) {
  const int ncol  = nb * 16 + (lane & 15);
  const int kbase = kb + ((lane >> 4) << 4);
  v16h bm;
#pragma unroll
  for (int i = 0; i < 16; ++i) bm[i] = (_Float16)W[(size_t)(kbase + i) * ldn + ncol];
  return bm;
}

// Transposed B: B[k][n] = W[n][k], with W row-major [n][ldk]. Used for backward (x @ W^T).
__device__ inline v16h glb_load_BT(const float* W, int ldk, int lane, int kb, int nb) {
  const int ncol  = nb * 16 + (lane & 15);
  const int kbase = kb + ((lane >> 4) << 4);
  v16h bm;
#pragma unroll
  for (int i = 0; i < 16; ++i) bm[i] = (_Float16)W[(size_t)ncol * ldk + (kbase + i)];
  return bm;
}

#define WMMA_F16(a, b, c) \
  __builtin_amdgcn_wmma_f32_16x16x32_f16(false, (a), false, (b), (short)0, (c), false, false)

#define V8F_ZERO {0.f, 0.f, 0.f, 0.f, 0.f, 0.f, 0.f, 0.f}

// ==================== kernel 0: zero outputs + stats ====================
__global__ void k_init(float* out, float* stats) {
  const int i = blockIdx.x * blockDim.x + threadIdx.x;
  if (i < OUT_TOTAL) out[i] = 0.f;
  if (i < 8) stats[i] = 0.f;
}

// ==================== kernel 1: per-pair features -> S, feat, type stats ====================
// One block per (b,n); 256 threads, one pair each.
__global__ __launch_bounds__(256) void k_features(
    const float* __restrict__ rvec, const int* __restrict__ itype,
    const float* __restrict__ cpar, float* __restrict__ featws,
    float* __restrict__ Sws, float* stats) {
  const int bn = blockIdx.x;
  const int n  = bn % N;
  const int m  = threadIdx.x;

  __shared__ float Ss[M1 * 4];
  __shared__ float red[2];
  if (m < M1 * 4) Ss[m] = 0.f;
  if (m < 2) red[m] = 0.f;
  __syncthreads();

  const int ti = itype[n];
  const float* rv = rvec + ((size_t)bn * M + m) * 3;
  __builtin_prefetch(rv, 0, 1);  // global_prefetch_b8
  const float x = rv[0], y = rv[1], z = rv[2];
  const float r = sqrtf(x * x + y * y + z * z);
  const bool valid = r > 1e-6f;
  const float rs = valid ? r : 1.f;
  const float vm = valid ? 1.f : 0.f;
  float u = 2.f * (rs - RMIN) / SPAN - 1.f;
  u = fminf(1.f, fmaxf(-1.f, u));
  const float rc = fminf(fmaxf(rs, RMIN), RMAX);
  const float fc = (valid && rs < RMAX) ? 0.5f * (cosf(PI_F * (rc - RMIN) / SPAN) + 1.f) : 0.f;

  float T[BETA];
  T[0] = 1.f; T[1] = u;
#pragma unroll
  for (int k = 2; k < BETA; ++k) T[k] = 2.f * u * T[k - 1] - T[k - 2];

  const int tj = m >> 7;  // neighbor-type block
  const float* crow = cpar + (size_t)(ti * NTYPES + tj) * M1 * BETA;
  const float inv_r = 1.f / rs;
  const float q0 = vm, q1 = x * inv_r * vm, q2 = y * inv_r * vm, q3 = z * inv_r * vm;

#pragma unroll
  for (int p = 0; p < M1; ++p) {
    float P = 0.f;
#pragma unroll
    for (int k = 0; k < BETA; ++k) P += crow[p * BETA + k] * T[k];
    const float g = P * fc;
    atomicAdd(&Ss[p * 4 + 0], g * q0);
    atomicAdd(&Ss[p * 4 + 1], g * q1);
    atomicAdd(&Ss[p * 4 + 2], g * q2);
    atomicAdd(&Ss[p * 4 + 3], g * q3);
  }
  __syncthreads();

  if (m < M1 * 4) {
    const float sv = Ss[m] * (1.f / (float)M);
    Sws[(size_t)bn * NFEAT + m] = sv;
    Ss[m] = sv;
  }
  __syncthreads();

  if (m < NFEAT) {
    const int p = m >> 2, q = m & 3;  // feat[p][q] = S[p] . S[q]
    float f = 0.f;
#pragma unroll
    for (int a = 0; a < 4; ++a) f += Ss[p * 4 + a] * Ss[q * 4 + a];
    featws[(size_t)bn * NFEAT + m] = f;
    atomicAdd(&red[0], f);
    atomicAdd(&red[1], f * f);
  }
  __syncthreads();
  if (m == 0) {
    atomicAdd(&stats[ti], red[0]);
    atomicAdd(&stats[2 + ti], red[1]);
  }
}

// ==================== kernel 2: finalize per-type mean/std ====================
__global__ void k_stats(float* stats) {
  const int t = threadIdx.x;
  if (t < NTYPES) {
    const float cnt = (float)(N / 2) * (float)B * (float)NFEAT;
    const float mu  = stats[t] / cnt;
    const float var = (stats[2 + t] - cnt * mu * mu) / (cnt - 1.f);
    stats[4 + t] = mu;
    stats[6 + t] = sqrtf(fmaxf(var, 1e-30f));
  }
}

// ==================== kernel 3: MLP forward (WMMA), Ei & Etot ====================
// One wave32 block per 16-atom tile (type-uniform: Imagetype_map is sorted, N/2 % 16 == 0).
__global__ __launch_bounds__(32) void k_mlp_fwd(
    const float* __restrict__ featws, const int* __restrict__ itype, const float* __restrict__ stats,
    const float* __restrict__ W0, const float* __restrict__ b0,
    const float* __restrict__ W1, const float* __restrict__ b1,
    const float* __restrict__ W2, const float* __restrict__ b2,
    const float* __restrict__ Wout, const float* __restrict__ bout,
    float* __restrict__ h0ws, float* __restrict__ t1ws, float* __restrict__ t2ws,
    float* __restrict__ out) {
  __shared__ float Xs[16 * H];
  __shared__ float Hs[16 * H];
  const int lane = threadIdx.x;
  const int atom0 = blockIdx.x * 16;
  const int b = atom0 / N, n0 = atom0 % N;
  const int t = itype[n0];
  const float mu = stats[4 + t], inv_sd = 1.f / stats[6 + t];

  const float* W0t = W0 + (size_t)t * NFEAT * H;  const float* b0t = b0 + t * H;
  const float* W1t = W1 + (size_t)t * H * H;      const float* b1t = b1 + t * H;
  const float* W2t = W2 + (size_t)t * H * H;      const float* b2t = b2 + t * H;
  const float* Wot = Wout + (size_t)t * H;        const float  eb  = bout[t];
  __builtin_prefetch(W1t + lane * 64, 0, 3);
  __builtin_prefetch(W2t + lane * 64, 0, 3);

#ifdef HAVE_ASYNC_LDS
  // Stage raw 16x64 feature tile straight into LDS via the async copy path
  // (global_load_async_to_lds_b128, tracked on ASYNCcnt), then normalize in LDS.
#pragma unroll
  for (int it = 0; it < 8; ++it) {
    const int idx = lane + it * 32;            // 0..255 vec4 groups
    const int a = idx >> 4, fg = (idx & 15) << 2;
    __builtin_amdgcn_global_load_async_to_lds_b128(
        GPTR(featws + (size_t)(atom0 + a) * NFEAT + fg), LPTR(&Xs[a * H + fg]), 0, 0);
  }
  __builtin_amdgcn_s_wait_asynccnt(0);
  __syncthreads();
  for (int idx = lane; idx < 16 * NFEAT; idx += 32) {
    const int a = idx / NFEAT, f = idx % NFEAT;
    Xs[a * H + f] = (Xs[a * H + f] - mu) * inv_sd;
  }
#else
  // normalized feature tile -> Xs[:, 0:64]
  for (int idx = lane; idx < 16 * NFEAT; idx += 32) {
    const int a = idx / NFEAT, f = idx % NFEAT;
    Xs[a * H + f] = (featws[(size_t)(atom0 + a) * NFEAT + f] - mu) * inv_sd;
  }
#endif
  __syncthreads();

  const int rbase = (lane >> 4) << 3;  // C/D row base (0 or 8)
  const int ncl = lane & 15;

  // layer0: h0 = tanh(X @ W0 + b0) -> Hs, h0ws
  for (int nb = 0; nb < 8; ++nb) {
    v8f c = V8F_ZERO;
    for (int kb = 0; kb < NFEAT; kb += 32)
      c = WMMA_F16(lds_load_A(Xs, H, lane, kb), glb_load_B(W0t, H, lane, kb, nb), c);
    const int col = nb * 16 + ncl;
#pragma unroll
    for (int i = 0; i < 8; ++i) {
      const int row = rbase + i;
      const float v = tanhf(c[i] + b0t[col]);
      Hs[row * H + col] = v;
      h0ws[(size_t)(atom0 + row) * H + col] = v;
    }
  }
  __syncthreads();

  // layer1: t1 = tanh(h0 @ W1 + b1); h1 = t1 + h0 -> Xs
  for (int nb = 0; nb < 8; ++nb) {
    v8f c = V8F_ZERO;
    for (int kb = 0; kb < H; kb += 32)
      c = WMMA_F16(lds_load_A(Hs, H, lane, kb), glb_load_B(W1t, H, lane, kb, nb), c);
    const int col = nb * 16 + ncl;
#pragma unroll
    for (int i = 0; i < 8; ++i) {
      const int row = rbase + i;
      const float tt = tanhf(c[i] + b1t[col]);
      t1ws[(size_t)(atom0 + row) * H + col] = tt;
      Xs[row * H + col] = tt + Hs[row * H + col];
    }
  }
  __syncthreads();

  // layer2: t2 = tanh(h1 @ W2 + b2); h2 = t2 + h1 -> Hs
  for (int nb = 0; nb < 8; ++nb) {
    v8f c = V8F_ZERO;
    for (int kb = 0; kb < H; kb += 32)
      c = WMMA_F16(lds_load_A(Xs, H, lane, kb), glb_load_B(W2t, H, lane, kb, nb), c);
    const int col = nb * 16 + ncl;
#pragma unroll
    for (int i = 0; i < 8; ++i) {
      const int row = rbase + i;
      const float tt = tanhf(c[i] + b2t[col]);
      t2ws[(size_t)(atom0 + row) * H + col] = tt;
      Hs[row * H + col] = tt + Xs[row * H + col];
    }
  }
  __syncthreads();

  // output head: e = h2 . Wout + bout  (split 128-dot across lane pairs, wave32 shuffle combine)
  const int a_ = lane & 15, half = lane >> 4;
  float part = 0.f;
  for (int cix = half * 64; cix < half * 64 + 64; ++cix) part += Hs[a_ * H + cix] * Wot[cix];
  part += __shfl_xor(part, 16, 32);
  if (lane < 16) {
    const float e = part + eb;
    out[OUT_EI + (size_t)b * N + (n0 + lane)] = e;
    atomicAdd(&out[OUT_ETOT + b], e);
  }
}

// ==================== kernel 4: MLP backward (WMMA, transposed weights) -> dE/dfeat ====================
__global__ __launch_bounds__(32) void k_mlp_bwd(
    const int* __restrict__ itype, const float* __restrict__ stats,
    const float* __restrict__ W0, const float* __restrict__ W1, const float* __restrict__ W2,
    const float* __restrict__ Wout,
    const float* __restrict__ h0ws, const float* __restrict__ t1ws, const float* __restrict__ t2ws,
    float* __restrict__ dfeatws) {
  __shared__ float Xs[16 * H];
  __shared__ float Hs[16 * H];
  const int lane = threadIdx.x;
  const int atom0 = blockIdx.x * 16;
  const int t = itype[atom0 % N];
  const float inv_sd = 1.f / stats[6 + t];
  const float* W0t = W0 + (size_t)t * NFEAT * H;
  const float* W1t = W1 + (size_t)t * H * H;
  const float* W2t = W2 + (size_t)t * H * H;
  const float* Wot = Wout + (size_t)t * H;
  const int rbase = (lane >> 4) << 3;
  const int ncl = lane & 15;

  // dt2 = (1 - t2^2) * dh2,  dh2[col] = Wout[col] (same for every atom)
  for (int idx = lane; idx < 16 * H; idx += 32) {
    const float tt = t2ws[(size_t)atom0 * H + idx];
    Xs[idx] = (1.f - tt * tt) * Wot[idx % H];
  }
  __syncthreads();

  // dh1 = dh2 + dt2 @ W2^T -> Hs
  for (int nb = 0; nb < 8; ++nb) {
    v8f c = V8F_ZERO;
    for (int kb = 0; kb < H; kb += 32)
      c = WMMA_F16(lds_load_A(Xs, H, lane, kb), glb_load_BT(W2t, H, lane, kb, nb), c);
    const int col = nb * 16 + ncl;
#pragma unroll
    for (int i = 0; i < 8; ++i) Hs[(rbase + i) * H + col] = c[i] + Wot[col];
  }
  __syncthreads();

  // dt1 = (1 - t1^2) * dh1 -> Xs
  for (int idx = lane; idx < 16 * H; idx += 32) {
    const float tt = t1ws[(size_t)atom0 * H + idx];
    Xs[idx] = (1.f - tt * tt) * Hs[idx];
  }
  __syncthreads();

  // dh0 = dh1 + dt1 @ W1^T -> Hs (in place)
  for (int nb = 0; nb < 8; ++nb) {
    v8f c = V8F_ZERO;
    for (int kb = 0; kb < H; kb += 32)
      c = WMMA_F16(lds_load_A(Xs, H, lane, kb), glb_load_BT(W1t, H, lane, kb, nb), c);
    const int col = nb * 16 + ncl;
#pragma unroll
    for (int i = 0; i < 8; ++i) {
      const int pos = (rbase + i) * H + col;
      Hs[pos] = c[i] + Hs[pos];
    }
  }
  __syncthreads();

  // dg0 = (1 - h0^2) * dh0 -> Xs
  for (int idx = lane; idx < 16 * H; idx += 32) {
    const float hh = h0ws[(size_t)atom0 * H + idx];
    Xs[idx] = (1.f - hh * hh) * Hs[idx];
  }
  __syncthreads();

  // dfeat = (dg0 @ W0^T) / std   (16 x 64)
  for (int nb = 0; nb < 4; ++nb) {
    v8f c = V8F_ZERO;
    for (int kb = 0; kb < H; kb += 32)
      c = WMMA_F16(lds_load_A(Xs, H, lane, kb), glb_load_BT(W0t, H, lane, kb, nb), c);
    const int col = nb * 16 + ncl;
#pragma unroll
    for (int i = 0; i < 8; ++i)
      dfeatws[(size_t)(atom0 + rbase + i) * NFEAT + col] = c[i] * inv_sd;
  }
}

// ==================== kernel 5: dE/dS from dE/dfeat (feat = S . S[:4]^T) ====================
__global__ __launch_bounds__(64) void k_dS(
    const float* __restrict__ dfeatws, const float* __restrict__ Sws, float* __restrict__ dSws) {
  const int bn = blockIdx.x, tid = threadIdx.x;
  __shared__ float dF[NFEAT], Ss[NFEAT];
  dF[tid] = dfeatws[(size_t)bn * NFEAT + tid];
  Ss[tid] = Sws[(size_t)bn * NFEAT + tid];
  __syncthreads();
  const int p = tid >> 2, a = tid & 3;
  float acc = 0.f;
#pragma unroll
  for (int q = 0; q < 4; ++q) acc += dF[p * 4 + q] * Ss[q * 4 + a];
  if (p < 4) {
#pragma unroll
    for (int pp = 0; pp < M1; ++pp) acc += dF[pp * 4 + p] * Ss[pp * 4 + a];
  }
  dSws[(size_t)bn * NFEAT + tid] = acc;
}

// ==================== kernel 6: per-pair backward; scatter forces & virial ====================
__global__ __launch_bounds__(256) void k_pair_bwd(
    const float* __restrict__ rvec, const int* __restrict__ itype,
    const float* __restrict__ cpar, const int* __restrict__ list_neigh,
    const float* __restrict__ dSws, float* __restrict__ out) {
  const int bn = blockIdx.x;
  const int b = bn / N, n = bn % N;
  const int m = threadIdx.x;

  __shared__ float dSs[NFEAT];
  __shared__ float Fc[3];
  __shared__ float Vir[9];
#ifdef HAVE_ASYNC_LDS
  if (m < 16)  // 16 lanes x b128 = 64 floats of dE/dS, async copy to LDS
    __builtin_amdgcn_global_load_async_to_lds_b128(
        GPTR(dSws + (size_t)bn * NFEAT + m * 4), LPTR(&dSs[m * 4]), 0, 0);
#else
  if (m < NFEAT) dSs[m] = dSws[(size_t)bn * NFEAT + m];
#endif
  if (m >= 64 && m < 67) Fc[m - 64] = 0.f;
  if (m >= 80 && m < 89) Vir[m - 80] = 0.f;
#ifdef HAVE_ASYNC_LDS
  __builtin_amdgcn_s_wait_asynccnt(0);
#endif
  __syncthreads();

  const int ti = itype[n];
  const float* rv = rvec + ((size_t)bn * M + m) * 3;
  __builtin_prefetch(rv, 0, 1);
  const float x = rv[0], y = rv[1], z = rv[2];
  const float r = sqrtf(x * x + y * y + z * z);
  const bool valid = r > 1e-6f;
  const float rs = valid ? r : 1.f;
  const float vm = valid ? 1.f : 0.f;
  const float u0 = 2.f * (rs - RMIN) / SPAN - 1.f;
  const float u = fminf(1.f, fmaxf(-1.f, u0));
  const float dudr = (u0 > -1.f && u0 < 1.f) ? 2.f / SPAN : 0.f;  // grad through clip
  const float rc = fminf(fmaxf(rs, RMIN), RMAX);
  const float fc = (valid && rs < RMAX) ? 0.5f * (cosf(PI_F * (rc - RMIN) / SPAN) + 1.f) : 0.f;
  const float dfc = (valid && rs < RMAX && rs > RMIN)
                        ? (-0.5f * PI_F / SPAN) * sinf(PI_F * (rc - RMIN) / SPAN)
                        : 0.f;

  float T[BETA], dT[BETA];
  T[0] = 1.f; T[1] = u; dT[0] = 0.f; dT[1] = 1.f;
#pragma unroll
  for (int k = 2; k < BETA; ++k) {
    T[k]  = 2.f * u * T[k - 1] - T[k - 2];
    dT[k] = 2.f * T[k - 1] + 2.f * u * dT[k - 1] - dT[k - 2];
  }

  const int tj = m >> 7;
  const float* crow = cpar + (size_t)(ti * NTYPES + tj) * M1 * BETA;
  const float inv_r = 1.f / rs;
  const float un[3] = {x * inv_r, y * inv_r, z * inv_r};
  const float q[4] = {vm, un[0] * vm, un[1] * vm, un[2] * vm};
  const float invM = 1.f / (float)M;

  float dEdr_g = 0.f;       // scalar dE/dr through g(r) path
  float dEq[4] = {0.f, 0.f, 0.f, 0.f};
#pragma unroll
  for (int p = 0; p < M1; ++p) {
    float P = 0.f, Q = 0.f;
#pragma unroll
    for (int k = 0; k < BETA; ++k) {
      const float ck = crow[p * BETA + k];
      P += ck * T[k];
      Q += ck * dT[k];
    }
    const float g = P * fc;
    const float dEdg = invM * (dSs[p * 4 + 0] * q[0] + dSs[p * 4 + 1] * q[1] +
                               dSs[p * 4 + 2] * q[2] + dSs[p * 4 + 3] * q[3]);
    dEdr_g += dEdg * (fc * Q * dudr + P * dfc);
    dEq[1] += dSs[p * 4 + 1] * g;
    dEq[2] += dSs[p * 4 + 2] * g;
    dEq[3] += dSs[p * 4 + 3] * g;
  }
  dEq[1] *= invM; dEq[2] *= invM; dEq[3] *= invM;

  // dE/drvec_c = dEdr_g * un_c + vm/r * sum_a dEq[a+1] * (delta_ac - un_a un_c)
  float dEdrv[3];
#pragma unroll
  for (int c3 = 0; c3 < 3; ++c3) {
    float s = 0.f;
#pragma unroll
    for (int a = 0; a < 3; ++a)
      s += dEq[a + 1] * (((a == c3) ? 1.f : 0.f) - un[a] * un[c3]);
    dEdrv[c3] = dEdr_g * un[c3] + vm * inv_r * s;
  }

  // center-atom force: +sum_m dEdr (block reduce, then one global atomic)
  atomicAdd(&Fc[0], dEdrv[0]);
  atomicAdd(&Fc[1], dEdrv[1]);
  atomicAdd(&Fc[2], dEdrv[2]);
  // virial: -sum rvec_a * dEdr_c
  const float rvv[3] = {x, y, z};
#pragma unroll
  for (int a = 0; a < 3; ++a)
#pragma unroll
    for (int c3 = 0; c3 < 3; ++c3) atomicAdd(&Vir[a * 3 + c3], -rvv[a] * dEdrv[c3]);
  // neighbor scatter (includes ghosts): -dEdr where list entry > 0
  const int nl = list_neigh[(((size_t)b * N + n) * NTYPES + tj) * MN + (m & (MN - 1))];
  if (nl > 0) {
    const size_t base = OUT_FORCE + ((size_t)b * (N + NGHOST) + (nl - 1)) * 3;
    atomicAdd(&out[base + 0], -dEdrv[0]);
    atomicAdd(&out[base + 1], -dEdrv[1]);
    atomicAdd(&out[base + 2], -dEdrv[2]);
  }
  __syncthreads();
  if (m < 3)
    atomicAdd(&out[OUT_FORCE + ((size_t)b * (N + NGHOST) + n) * 3 + m], Fc[m]);
  if (m >= 32 && m < 41)
    atomicAdd(&out[OUT_VIR + b * 9 + (m - 32)], Vir[m - 32]);
}

// ==================== host-side launch ====================
extern "C" void kernel_launch(void* const* d_in, const int* in_sizes, int n_in,
                              void* d_out, int out_size, void* d_ws, size_t ws_size,
                              hipStream_t stream) {
  (void)in_sizes; (void)n_in; (void)out_size; (void)ws_size;
  const int*   list_neigh = (const int*)d_in[0];
  const int*   itype      = (const int*)d_in[1];
  const float* rvec       = (const float*)d_in[2];
  const float* cpar       = (const float*)d_in[3];
  const float* W0   = (const float*)d_in[4];
  const float* b0   = (const float*)d_in[5];
  const float* W1   = (const float*)d_in[6];
  const float* b1   = (const float*)d_in[7];
  const float* W2   = (const float*)d_in[8];
  const float* b2   = (const float*)d_in[9];
  const float* Wout = (const float*)d_in[10];
  const float* bout = (const float*)d_in[11];

  float* out = (float*)d_out;
  float* ws  = (float*)d_ws;
  float* featws  = ws + WS_FEAT;
  float* Sws     = ws + WS_S;
  float* h0ws    = ws + WS_H0;
  float* t1ws    = ws + WS_T1;
  float* t2ws    = ws + WS_T2;
  float* dfeatws = ws + WS_DFEAT;
  float* dSws    = ws + WS_DS;
  float* stats   = ws + WS_STATS;

  k_init<<<(OUT_TOTAL + 255) / 256, 256, 0, stream>>>(out, stats);
  k_features<<<B * N, 256, 0, stream>>>(rvec, itype, cpar, featws, Sws, stats);
  k_stats<<<1, 32, 0, stream>>>(stats);
  k_mlp_fwd<<<B * N / 16, 32, 0, stream>>>(featws, itype, stats, W0, b0, W1, b1, W2, b2,
                                           Wout, bout, h0ws, t1ws, t2ws, out);
  k_mlp_bwd<<<B * N / 16, 32, 0, stream>>>(itype, stats, W0, W1, W2, Wout,
                                           h0ws, t1ws, t2ws, dfeatws);
  k_dS<<<B * N, 64, 0, stream>>>(dfeatws, Sws, dSws);
  k_pair_bwd<<<B * N, 256, 0, stream>>>(rvec, itype, cpar, list_neigh, dSws, out);
}
